// FinalAggregatorBatch_28054726377752
// MI455X (gfx1250) — compile-verified, matrix operand order checked by reading, not run on hardware
//
#include <hip/hip_runtime.h>
#include <hip/hip_bf16.h>
#include <cstdint>
#include <cstddef>

// ---------------------------------------------------------------------------
// Constants from the reference
// ---------------------------------------------------------------------------
#define C_DIM 512
#define NHOP  3
#define BSUB  8
#define LNEPS 1e-5f

typedef __bf16 bf16_t;
typedef __attribute__((ext_vector_type(16))) __bf16 v16bf;
typedef __attribute__((ext_vector_type(8)))  float  v8f;

// ---------------------------------------------------------------------------
// WMMA bf16 GEMM:  Y[M,N] = A[M,K] @ W[N,K]^T + bias     (K,N = 512 here)
// Block: 256 threads (8 wave32 arranged 4x2), block tile 128x64, K-step 32.
// Each wave owns a 32x32 output tile: 2 A-frags x 2 B-frags -> 4 v_wmma per
// K-step with 8 ds_load_b128 (2:1 ds:wmma, double the ILP of a 16x16 wave
// tile -- helps cover the 5-cycle WMMA->WMMA accumulate latency).
// A/W are fp32 in global; converted to bf16 while staging through LDS.
// ---------------------------------------------------------------------------
#define TM 128
#define TN 64
#define TK 32
#define LDSS 48   // LDS row stride (bf16 elems) = 96B: keeps 16B chunks aligned

__global__ __launch_bounds__(256) void wmma_gemm_bias_kernel(
    const float* __restrict__ A, const float* __restrict__ W,
    const float* __restrict__ bias, float* __restrict__ Y,
    int M, int K, int N)
{
  __shared__ bf16_t As[TM * LDSS];   // 12 KB
  __shared__ bf16_t Bs[TN * LDSS];   //  6 KB

  const int tid  = threadIdx.x;
  const int lane = tid & 31;
  const int wave = tid >> 5;      // 0..7
  const int mw   = wave & 3;      // wave M tile: offsets mw*32
  const int nw   = wave >> 2;     // wave N tile: offsets nw*32
  const int m16  = lane & 15;
  const int half = lane >> 4;

  const int rowBase = blockIdx.y * TM;
  const int colBase = blockIdx.x * TN;

  // cooperative tile staging: 256 threads, 8 fp32 per thread per 64-row panel
  const int lr = tid >> 2;          // 0..63
  const int lc = (tid & 3) * 8;     // 0,8,16,24

  v8f c00 = {0.f,0.f,0.f,0.f,0.f,0.f,0.f,0.f};
  v8f c01 = {0.f,0.f,0.f,0.f,0.f,0.f,0.f,0.f};
  v8f c10 = {0.f,0.f,0.f,0.f,0.f,0.f,0.f,0.f};
  v8f c11 = {0.f,0.f,0.f,0.f,0.f,0.f,0.f,0.f};

  const float* ga0 = A + (size_t)(rowBase + lr) * K + lc;        // A rows 0..63
  const float* ga1 = A + (size_t)(rowBase + 64 + lr) * K + lc;   // A rows 64..127
  const float* gb  = W + (size_t)(colBase + lr) * K + lc;        // W rows 0..63
  bf16_t* sa0 = As + lr * LDSS + lc;
  bf16_t* sa1 = As + (64 + lr) * LDSS + lc;
  bf16_t* sb  = Bs + lr * LDSS + lc;

  for (int k0 = 0; k0 < K; k0 += TK) {
#pragma unroll
    for (int i = 0; i < 8; ++i) sa0[i] = (bf16_t)ga0[k0 + i];
#pragma unroll
    for (int i = 0; i < 8; ++i) sa1[i] = (bf16_t)ga1[k0 + i];
#pragma unroll
    for (int i = 0; i < 8; ++i) sb[i]  = (bf16_t)gb[k0 + i];
    if (k0 + TK < K) {            // near-cache prefetch of next K tile
      __builtin_prefetch(ga0 + k0 + TK, 0, 3);
      __builtin_prefetch(ga1 + k0 + TK, 0, 3);
      __builtin_prefetch(gb  + k0 + TK, 0, 3);
    }
    __syncthreads();

    // wave32 WMMA fragment layout (ISA 7.12.2):
    //  lane m holds K = {half*8..+7} and {16+half*8..+7} of row m
    const bf16_t* ap0 = As + (mw * 32 + m16) * LDSS;
    const bf16_t* ap1 = ap0 + 16 * LDSS;
    const bf16_t* bp0 = Bs + (nw * 32 + m16) * LDSS;
    const bf16_t* bp1 = bp0 + 16 * LDSS;
    const int klo = half * 8;
    v16bf a0, a1, b0, b1;
#pragma unroll
    for (int i = 0; i < 8; ++i) {
      a0[i]      = ap0[klo + i];
      a0[i + 8]  = ap0[16 + klo + i];
      a1[i]      = ap1[klo + i];
      a1[i + 8]  = ap1[16 + klo + i];
      b0[i]      = bp0[klo + i];
      b0[i + 8]  = bp0[16 + klo + i];
      b1[i]      = bp1[klo + i];
      b1[i + 8]  = bp1[16 + klo + i];
    }
    c00 = __builtin_amdgcn_wmma_f32_16x16x32_bf16(false, a0, false, b0,
                                                  (short)0, c00, false, false);
    c01 = __builtin_amdgcn_wmma_f32_16x16x32_bf16(false, a0, false, b1,
                                                  (short)0, c01, false, false);
    c10 = __builtin_amdgcn_wmma_f32_16x16x32_bf16(false, a1, false, b0,
                                                  (short)0, c10, false, false);
    c11 = __builtin_amdgcn_wmma_f32_16x16x32_bf16(false, a1, false, b1,
                                                  (short)0, c11, false, false);
    __syncthreads();
  }

  // C/D layout: reg r -> row 8*half + r, col = lane&15
  const int col0 = colBase + nw * 32 + m16;
  const int col1 = col0 + 16;
  const float bb0 = bias[col0];
  const float bb1 = bias[col1];
#pragma unroll
  for (int r = 0; r < 8; ++r) {
    const int row0 = rowBase + mw * 32 + half * 8 + r;
    const int row1 = row0 + 16;
    Y[(size_t)row0 * N + col0] = c00[r] + bb0;
    Y[(size_t)row0 * N + col1] = c01[r] + bb1;
    Y[(size_t)row1 * N + col0] = c10[r] + bb0;
    Y[(size_t)row1 * N + col1] = c11[r] + bb1;
  }
  (void)M;
}

// ---------------------------------------------------------------------------
// Input projection: X = relu(Xraw @ W^T + b), small K (8 or 10)
// ---------------------------------------------------------------------------
__global__ void proj_relu_kernel(const float* __restrict__ Xr,
                                 const float* __restrict__ W,
                                 const float* __restrict__ b,
                                 float* __restrict__ X,
                                 long long R, int K)
{
  long long i = (long long)blockIdx.x * blockDim.x + threadIdx.x;
  if (i >= R * C_DIM) return;
  long long r = i / C_DIM;
  int c = (int)(i % C_DIM);
  const float* xr = Xr + r * K;
  const float* wr = W + (size_t)c * K;
  float acc = b[c];
  for (int k = 0; k < K; ++k) acc = fmaf(xr[k], wr[k], acc);
  X[i] = acc > 0.f ? acc : 0.f;
}

// ---------------------------------------------------------------------------
// Scatter segment-max over edges.  Values are >=0 (post-ReLU), dest zeroed,
// so uint atomicMax on the IEEE bits == float max with fill 0.
// ---------------------------------------------------------------------------
__global__ void seg_max_edges_kernel(const float* __restrict__ X,
                                     const int* __restrict__ ei,
                                     int G, int E, int N,
                                     float* __restrict__ agg)
{
  long long i = (long long)blockIdx.x * blockDim.x + threadIdx.x;
  long long total = (long long)G * E * C_DIM;
  if (i >= total) return;
  int c = (int)(i % C_DIM);
  long long ge = i / C_DIM;
  int e = (int)(ge % E);
  int g = (int)(ge / E);
  const int* eig = ei + (size_t)g * 2 * E;
  int src = eig[e];
  int dst = eig[E + e];
  float v = X[((size_t)g * N + src) * C_DIM + c];
  atomicMax((unsigned int*)(agg + ((size_t)g * N + dst) * C_DIM + c),
            __float_as_uint(v));
}

// Polygon-masked (edge_attr == 3) segment-max + has_poly flag
__global__ void poly_seg_max_kernel(const float* __restrict__ NF,
                                    const int* __restrict__ ei,
                                    const int* __restrict__ ea,
                                    int G, int E, int N,
                                    float* __restrict__ agg,
                                    int* __restrict__ haspoly)
{
  long long i = (long long)blockIdx.x * blockDim.x + threadIdx.x;
  long long total = (long long)G * E * C_DIM;
  if (i >= total) return;
  int c = (int)(i % C_DIM);
  long long ge = i / C_DIM;
  int e = (int)(ge % E);
  int g = (int)(ge / E);
  if (ea[(size_t)g * E + e] != 3) return;
  const int* eig = ei + (size_t)g * 2 * E;
  int src = eig[e];
  int dst = eig[E + e];
  if (c == 0) haspoly[(size_t)g * N + dst] = 1;   // benign same-value race
  float v = NF[((size_t)g * N + src) * C_DIM + c];
  atomicMax((unsigned int*)(agg + ((size_t)g * N + dst) * C_DIM + c),
            __float_as_uint(v));
}

// Weighted scatter-add (large block): agg[dst] += x[src] * ea[e]
__global__ void wsum_edges_kernel(const float* __restrict__ X,
                                  const int* __restrict__ ei,
                                  const float* __restrict__ ea,
                                  int E, float* __restrict__ agg)
{
  long long i = (long long)blockIdx.x * blockDim.x + threadIdx.x;
  long long total = (long long)E * C_DIM;
  if (i >= total) return;
  int c = (int)(i % C_DIM);
  int e = (int)(i / C_DIM);
  int src = ei[e];
  int dst = ei[E + e];
  atomicAdd(agg + (size_t)dst * C_DIM + c,
            X[(size_t)src * C_DIM + c] * ea[e]);
}

// ---------------------------------------------------------------------------
// Per-segment sum / sum-sq for layernorm stats (segments are contiguous rows)
// grid = (chunks, nsegs)
// ---------------------------------------------------------------------------
__global__ void reduce_stats_kernel(const float* __restrict__ Y,
                                    long long segElems,
                                    float* __restrict__ stats)
{
  int seg = blockIdx.y;
  const float* base = Y + (size_t)seg * segElems;
  long long start = (long long)blockIdx.x * (blockDim.x * 8) + threadIdx.x;
  float s = 0.f, ss = 0.f;
#pragma unroll
  for (int j = 0; j < 8; ++j) {
    long long idx = start + (long long)j * blockDim.x;
    if (idx < segElems) { float v = base[idx]; s += v; ss = fmaf(v, v, ss); }
  }
  __shared__ float sh0[256], sh1[256];
  sh0[threadIdx.x] = s; sh1[threadIdx.x] = ss;
  __syncthreads();
  for (int o = 128; o > 0; o >>= 1) {
    if ((int)threadIdx.x < o) {
      sh0[threadIdx.x] += sh0[threadIdx.x + o];
      sh1[threadIdx.x] += sh1[threadIdx.x + o];
    }
    __syncthreads();
  }
  if (threadIdx.x == 0) {
    atomicAdd(&stats[2 * seg],     sh0[0]);
    atomicAdd(&stats[2 * seg + 1], sh1[0]);
  }
}

// x = relu(ln(Y)+x) ; fused += x * sigmoid(gate)
__global__ void hop_post_gated_kernel(const float* __restrict__ Y,
                                      float* __restrict__ X,
                                      float* __restrict__ FUSED,
                                      const float* __restrict__ stats,
                                      long long segElems,
                                      const float* __restrict__ lnw,
                                      const float* __restrict__ lnb,
                                      const float* __restrict__ gate,
                                      long long total)
{
  long long i = (long long)blockIdx.x * blockDim.x + threadIdx.x;
  if (i >= total) return;
  int seg = (int)(i / segElems);
  int c = (int)(i % C_DIM);
  float inv = 1.f / (float)segElems;
  float mu  = stats[2 * seg] * inv;
  float var = stats[2 * seg + 1] * inv - mu * mu;
  float y = (Y[i] - mu) * rsqrtf(var + LNEPS) * lnw[c] + lnb[c];
  float xn = fmaxf(y + X[i], 0.f);
  X[i] = xn;
  float sig = 1.f / (1.f + expf(-gate[c]));
  FUSED[i] += xn * sig;
}

// out = relu(ln(Y) + res)
__global__ void hop_post_plain_kernel(const float* __restrict__ Y,
                                      const float* __restrict__ RES,
                                      float* __restrict__ OUT,
                                      const float* __restrict__ stats,
                                      long long segElems,
                                      const float* __restrict__ lnw,
                                      const float* __restrict__ lnb,
                                      long long total)
{
  long long i = (long long)blockIdx.x * blockDim.x + threadIdx.x;
  if (i >= total) return;
  int seg = (int)(i / segElems);
  int c = (int)(i % C_DIM);
  float inv = 1.f / (float)segElems;
  float mu  = stats[2 * seg] * inv;
  float var = stats[2 * seg + 1] * inv - mu * mu;
  float y = (Y[i] - mu) * rsqrtf(var + LNEPS) * lnw[c] + lnb[c];
  OUT[i] = fmaxf(y + RES[i], 0.f);
}

// ---------------------------------------------------------------------------
// Ring pooling: max over ring positions g of max(step2, remain) per (b,c)
// ---------------------------------------------------------------------------
__global__ void ring_pool_kernel(const float* __restrict__ SUB,
                                 const float* __restrict__ NF,
                                 const int* __restrict__ haspoly,
                                 const float* __restrict__ Xraw,
                                 int G, int inK, int N, int npg,
                                 float* __restrict__ out, int colOff)
{
  int i = blockIdx.x * blockDim.x + threadIdx.x;
  if (i >= BSUB * C_DIM) return;
  int b = i / C_DIM;
  int c = i % C_DIM;
  float best = -3.0e38f;
  for (int g = 0; g < G; ++g) {
    float s2 = -3.0e38f, rm = -3.0e38f;
    const size_t nb = (size_t)g * N + (size_t)b * npg;
    for (int n = 0; n < npg; ++n) {
      size_t node = nb + n;
      if (haspoly[node])                   s2 = fmaxf(s2, SUB[node * C_DIM + c]);
      if (Xraw[node * inK] <= 0.1f)        rm = fmaxf(rm, NF[node * C_DIM + c]);
    }
    if (s2 < -1.0e30f) s2 = 1e-4f;
    if (rm < -1.0e30f) rm = 1e-4f;
    best = fmaxf(best, fmaxf(s2, rm));
  }
  out[(size_t)b * (4 * C_DIM) + colOff + c] = best;
}

// global_max_pool over contiguous subgraphs (large block)
__global__ void seg_max_pool_kernel(const float* __restrict__ X, int rps,
                                    float* __restrict__ pooled)
{
  int i = blockIdx.x * blockDim.x + threadIdx.x;
  if (i >= BSUB * C_DIM) return;
  int b = i / C_DIM;
  int c = i % C_DIM;
  float m = -3.0e38f;
  for (int r = 0; r < rps; ++r)
    m = fmaxf(m, X[((size_t)b * rps + r) * C_DIM + c]);
  pooled[(size_t)b * C_DIM + c] = m;
}

__global__ void pool_linear_relu_kernel(const float* __restrict__ P,
                                        const float* __restrict__ W,
                                        const float* __restrict__ bias,
                                        float* __restrict__ out, int colOff)
{
  int i = blockIdx.x * blockDim.x + threadIdx.x;
  if (i >= BSUB * C_DIM) return;
  int b = i / C_DIM;
  int c = i % C_DIM;
  float acc = bias[c];
  const float* pr = P + (size_t)b * C_DIM;
  const float* wr = W + (size_t)c * C_DIM;
  for (int k = 0; k < C_DIM; ++k) acc = fmaf(pr[k], wr[k], acc);
  out[(size_t)b * (4 * C_DIM) + colOff + c] = fmaxf(acc, 0.f);
}

// ---------------------------------------------------------------------------
// MLP: tiny (batch 8) — plain kernels
// ---------------------------------------------------------------------------
__global__ void linear_kernel(const float* __restrict__ X,
                              const float* __restrict__ W,
                              const float* __restrict__ bias,
                              float* __restrict__ OUT,
                              int R, int Kin, int O)
{
  int i = blockIdx.x * blockDim.x + threadIdx.x;
  if (i >= R * O) return;
  int r = i / O;
  int o = i % O;
  float acc = bias[o];
  const float* xr = X + (size_t)r * Kin;
  const float* wr = W + (size_t)o * Kin;
  for (int k = 0; k < Kin; ++k) acc = fmaf(xr[k], wr[k], acc);
  OUT[i] = acc;
}

__global__ void rowln_relu_kernel(float* __restrict__ X,
                                  const float* __restrict__ w,
                                  const float* __restrict__ b, int D)
{
  int row = blockIdx.x;
  float* xr = X + (size_t)row * D;
  float s = 0.f, ss = 0.f;
  for (int d = threadIdx.x; d < D; d += blockDim.x) {
    float v = xr[d]; s += v; ss = fmaf(v, v, ss);
  }
  __shared__ float sh0[256], sh1[256];
  sh0[threadIdx.x] = s; sh1[threadIdx.x] = ss;
  __syncthreads();
  for (int o = 128; o > 0; o >>= 1) {
    if ((int)threadIdx.x < o) {
      sh0[threadIdx.x] += sh0[threadIdx.x + o];
      sh1[threadIdx.x] += sh1[threadIdx.x + o];
    }
    __syncthreads();
  }
  __shared__ float smu, sinv;
  if (threadIdx.x == 0) {
    float mu = sh0[0] / (float)D;
    float var = sh1[0] / (float)D - mu * mu;
    smu = mu; sinv = rsqrtf(var + LNEPS);
  }
  __syncthreads();
  for (int d = threadIdx.x; d < D; d += blockDim.x) {
    float v = (xr[d] - smu) * sinv * w[d] + b[d];
    xr[d] = fmaxf(v, 0.f);
  }
}

// ---------------------------------------------------------------------------
// Host driver
// ---------------------------------------------------------------------------
static inline unsigned cdiv_u(long long a, long long b) {
  return (unsigned)((a + b - 1) / b);
}

extern "C" void kernel_launch(void* const* d_in, const int* in_sizes, int n_in,
                              void* d_out, int out_size, void* d_ws, size_t ws_size,
                              hipStream_t stream)
{
  (void)in_sizes; (void)n_in; (void)out_size; (void)ws_size;
  const int C = C_DIM;
  const int TPB = 256;

  // ---- workspace carve-up (max rows = 16*1024 for the outer ring) ----
  char* base = (char*)d_ws;
  size_t off = 0;
  auto walloc = [&](size_t bytes) -> void* {
    void* p = base + off;
    off = (off + bytes + 255) & ~(size_t)255;
    return p;
  };
  const size_t MAXR = 16384;
  float* X     = (float*)walloc(MAXR * C * sizeof(float));
  float* FUSED = (float*)walloc(MAXR * C * sizeof(float));
  float* AGG   = (float*)walloc(MAXR * C * sizeof(float));
  float* Yb    = (float*)walloc(MAXR * C * sizeof(float));
  float* STATS = (float*)walloc(256 * 2 * sizeof(float));
  int*   HP    = (int*)  walloc(MAXR * sizeof(int));
  float* CONC  = (float*)walloc(8 * 4 * C * sizeof(float));
  float* T0    = (float*)walloc(8 * 4 * C * sizeof(float));
  float* T1    = (float*)walloc(8 * 4 * C * sizeof(float));
  float* POOL  = (float*)walloc(8 * C * sizeof(float));
  float* SUB   = AGG;   // agg buffer is free after the final-MPN GEMM

  // ---- one ring level (batched over its G graphs) ----
  auto run_small = [&](int G, const float* xr, const int* ei, const int* ea,
                       int pbase, int colOff) {
    const int N = 1024, E = 4096, K = 10;
    const long long R = (long long)G * N;
    const float* pW  = (const float*)d_in[pbase + 0];
    const float* pb  = (const float*)d_in[pbase + 1];
    const float* hW  = (const float*)d_in[pbase + 2];
    const float* hb  = (const float*)d_in[pbase + 3];
    const float* lw  = (const float*)d_in[pbase + 4];
    const float* lb  = (const float*)d_in[pbase + 5];
    const float* gt  = (const float*)d_in[pbase + 6];
    const float* fW  = (const float*)d_in[pbase + 7];
    const float* fb  = (const float*)d_in[pbase + 8];
    const float* flw = (const float*)d_in[pbase + 9];
    const float* flb = (const float*)d_in[pbase + 10];

    proj_relu_kernel<<<cdiv_u(R * C, TPB), TPB, 0, stream>>>(xr, pW, pb, X, R, K);
    hipMemsetAsync(FUSED, 0, R * C * sizeof(float), stream);

    const dim3 ggrid(C / TN, (unsigned)(R / TM));
    const long long total = R * C;
    const long long segE = (long long)N * C;          // per-graph LN

    for (int l = 0; l < NHOP; ++l) {
      hipMemsetAsync(AGG, 0, R * C * sizeof(float), stream);
      seg_max_edges_kernel<<<cdiv_u((long long)G * E * C, TPB), TPB, 0, stream>>>(
          X, ei, G, E, N, AGG);
      wmma_gemm_bias_kernel<<<ggrid, TPB, 0, stream>>>(
          AGG, hW + (size_t)l * C * C, hb + (size_t)l * C, Yb, (int)R, C, C);
      hipMemsetAsync(STATS, 0, G * 2 * sizeof(float), stream);
      reduce_stats_kernel<<<dim3(cdiv_u(segE, TPB * 8), G), TPB, 0, stream>>>(
          Yb, segE, STATS);
      hop_post_gated_kernel<<<cdiv_u(total, TPB), TPB, 0, stream>>>(
          Yb, X, FUSED, STATS, segE,
          lw + (size_t)l * C, lb + (size_t)l * C, gt + (size_t)l * C, total);
    }

    // final MPN on polygon edges (per-subgraph LN, 128 nodes/subgraph)
    hipMemsetAsync(AGG, 0, R * C * sizeof(float), stream);
    hipMemsetAsync(HP, 0, R * sizeof(int), stream);
    poly_seg_max_kernel<<<cdiv_u((long long)G * E * C, TPB), TPB, 0, stream>>>(
        FUSED, ei, ea, G, E, N, AGG, HP);
    wmma_gemm_bias_kernel<<<ggrid, TPB, 0, stream>>>(AGG, fW, fb, Yb, (int)R, C, C);
    const long long segF = 128LL * C;
    hipMemsetAsync(STATS, 0, G * BSUB * 2 * sizeof(float), stream);
    reduce_stats_kernel<<<dim3(cdiv_u(segF, TPB * 8), G * BSUB), TPB, 0, stream>>>(
        Yb, segF, STATS);
    hop_post_plain_kernel<<<cdiv_u(total, TPB), TPB, 0, stream>>>(
        Yb, FUSED, SUB, STATS, segF, flw, flb, total);
    ring_pool_kernel<<<cdiv_u(BSUB * C, TPB), TPB, 0, stream>>>(
        SUB, FUSED, HP, xr, G, K, N, 128, CONC, colOff);
  };

  run_small(16, (const float*)d_in[0], (const int*)d_in[1],
            (const int*)d_in[2], 16, 0);
  run_small(8,  (const float*)d_in[4], (const int*)d_in[5],
            (const int*)d_in[6], 27, C);
  run_small(1,  (const float*)d_in[8], (const int*)d_in[9],
            (const int*)d_in[10], 38, 2 * C);

  // ---- large block ----
  {
    const int N = 8192, E = 32768, K = 8;
    const long long R = N;
    const float* xr = (const float*)d_in[12];
    const int*   ei = (const int*)d_in[13];
    const float* ea = (const float*)d_in[14];
    const int pbase = 49;
    const float* pW = (const float*)d_in[pbase + 0];
    const float* pb = (const float*)d_in[pbase + 1];
    const float* hW = (const float*)d_in[pbase + 2];
    const float* hb = (const float*)d_in[pbase + 3];
    const float* lw = (const float*)d_in[pbase + 4];
    const float* lb = (const float*)d_in[pbase + 5];
    const float* oW = (const float*)d_in[pbase + 6];
    const float* ob = (const float*)d_in[pbase + 7];

    proj_relu_kernel<<<cdiv_u(R * C, TPB), TPB, 0, stream>>>(xr, pW, pb, X, R, K);
    const dim3 ggrid(C / TN, (unsigned)(R / TM));
    const long long total = R * C;
    const long long segE = total;                     // one graph, global LN
    for (int l = 0; l < NHOP; ++l) {
      hipMemsetAsync(AGG, 0, R * C * sizeof(float), stream);
      wsum_edges_kernel<<<cdiv_u((long long)E * C, TPB), TPB, 0, stream>>>(
          X, ei, ea, E, AGG);
      wmma_gemm_bias_kernel<<<ggrid, TPB, 0, stream>>>(
          AGG, hW + (size_t)l * C * C, hb + (size_t)l * C, Yb, (int)R, C, C);
      hipMemsetAsync(STATS, 0, 2 * sizeof(float), stream);
      reduce_stats_kernel<<<dim3(cdiv_u(segE, TPB * 8), 1), TPB, 0, stream>>>(
          Yb, segE, STATS);
      hop_post_plain_kernel<<<cdiv_u(total, TPB), TPB, 0, stream>>>(
          Yb, X, X, STATS, segE, lw + (size_t)l * C, lb + (size_t)l * C, total);
    }
    seg_max_pool_kernel<<<cdiv_u(BSUB * C, TPB), TPB, 0, stream>>>(X, 1024, POOL);
    pool_linear_relu_kernel<<<cdiv_u(BSUB * C, TPB), TPB, 0, stream>>>(
        POOL, oW, ob, CONC, 3 * C);
  }

  // ---- final MLP: [8,2048] -> ... -> [8,2] ----
  {
    const int dims[7] = {2048, 2048, 1024, 512, 256, 128, 2};
    const float* in = CONC;
    float* bufs[2] = {T0, T1};
    for (int i = 0; i < 5; ++i) {
      float* out = bufs[i & 1];
      linear_kernel<<<cdiv_u(8LL * dims[i + 1], TPB), TPB, 0, stream>>>(
          in, (const float*)d_in[57 + i], (const float*)d_in[63 + i],
          out, 8, dims[i], dims[i + 1]);
      rowln_relu_kernel<<<8, TPB, 0, stream>>>(
          out, (const float*)d_in[69 + i], (const float*)d_in[74 + i], dims[i + 1]);
      in = out;
    }
    linear_kernel<<<1, 64, 0, stream>>>(
        in, (const float*)d_in[62], (const float*)d_in[68],
        (float*)d_out, 8, 128, 2);
  }
}
